// VectorQuantizer_37443524887278
// MI455X (gfx1250) — matrix operand order repaired and checked by reading, not compiled, not run on hardware
//
#include <hip/hip_runtime.h>
#include <hip/hip_bf16.h>

typedef __attribute__((ext_vector_type(16))) _Float16 v16h;
typedef __attribute__((ext_vector_type(8)))  _Float16 v8h;
typedef __attribute__((ext_vector_type(4)))  _Float16 v4h;
typedef __attribute__((ext_vector_type(8)))  float    v8f;

#define VQ_B   64
#define VQ_C   128
#define VQ_HW  4096          // H*W
#define VQ_N   262144        // B*H*W
#define VQ_K   512           // codebook entries
#define MTILES 2             // 16-row A tiles per wave (register blocking on M)
#define ROWS_PER_WG (8 * 16 * MTILES)      // 256
#define NUM_WGS (VQ_N / ROWS_PER_WG)       // 1024
#define NZ 33554432          // B*C*H*W
#define EH_STRIDE 132        // padded halves per codebook row (264B, 8B aligned)

// LDS layout (bytes)
#define LDS_EH_OFF   0
#define LDS_EL_OFF   (VQ_K * EH_STRIDE * 2)                  // 135168
#define LDS_EN_OFF   (LDS_EL_OFF + VQ_K * EH_STRIDE * 2)     // 270336
#define LDS_IDX_OFF  (LDS_EN_OFF + VQ_K * 4)                 // 272384
#define LDS_RED_OFF  (LDS_IDX_OFF + ROWS_PER_WG * 4)         // 273408
#define LDS_BYTES    (LDS_RED_OFF + 8 * 4)                   // 273440

static __device__ inline v16h load_v16h(const _Float16* p) {
    const v4h* q = (const v4h*)p;          // 8B aligned by construction
    v4h a = q[0], b = q[1], c = q[2], d = q[3];
    v8h ab = __builtin_shufflevector(a, b, 0,1,2,3,4,5,6,7);
    v8h cd = __builtin_shufflevector(c, d, 0,1,2,3,4,5,6,7);
    return __builtin_shufflevector(ab, cd, 0,1,2,3,4,5,6,7,8,9,10,11,12,13,14,15);
}

__global__ __launch_bounds__(256)
void vq_phase1(const float* __restrict__ z_e, const float* __restrict__ emb,
               float* __restrict__ zq_out, float* __restrict__ idx_out,
               float* __restrict__ partial) {
    extern __shared__ char smem[];
    _Float16* eh    = (_Float16*)(smem + LDS_EH_OFF);
    _Float16* el    = (_Float16*)(smem + LDS_EL_OFF);
    float*    enorm = (float*)   (smem + LDS_EN_OFF);
    int*      idxs  = (int*)     (smem + LDS_IDX_OFF);
    float*    red   = (float*)   (smem + LDS_RED_OFF);

    const int tid = threadIdx.x;

    // ---- Stage codebook into LDS as f16 hi/lo (coalesced) ----
    for (int i = tid; i < VQ_K * VQ_C; i += 256) {
        float x = emb[i];
        _Float16 h = (_Float16)x;
        _Float16 l = (_Float16)(x - (float)h);
        int r = i >> 7, c = i & 127;
        eh[r * EH_STRIDE + c] = h;
        el[r * EH_STRIDE + c] = l;
    }
    // ---- ||e_k||^2 in fp32, fixed-order serial per row (deterministic) ----
    for (int r = tid; r < VQ_K; r += 256) {
        float s = 0.f;
        const float* er = emb + r * VQ_C;
        for (int c = 0; c < VQ_C; ++c) { float x = er[c]; s += x * x; }
        enorm[r] = s;
    }
    __syncthreads();

    const int wave = tid >> 5;
    const int lane = tid & 31;
    const int g    = lane >> 4;   // half-wave group
    const int m    = lane & 15;   // A row within tile / B column within tile

    // ---- Load MTILES A tiles (16 rows x 128 ch each) into registers, f16 hi/lo ----
    v16h ah[MTILES][4], al[MTILES][4];
#pragma unroll
    for (int u = 0; u < MTILES; ++u) {
        const int n   = blockIdx.x * ROWS_PER_WG + wave * (16 * MTILES) + u * 16 + m;
        const int bb  = n >> 12;
        const int pos = n & 4095;
        const float* zb = z_e + ((size_t)bb << 19) + pos;
#pragma unroll
        for (int cc = 0; cc < 4; ++cc) {
            v16h th = {}, tl = {};
#pragma unroll
            for (int e = 0; e < 16; ++e) {
                int kk = g * 16 + e;
                int ch = cc * 32 + kk;
                float x = zb[(size_t)ch << 12];
                _Float16 h = (_Float16)x;
                th[e] = h;
                tl[e] = (_Float16)(x - (float)h);
            }
            ah[u][cc] = th; al[u][cc] = tl;
        }
    }

    // ---- 32 column tiles: dist = ||e||^2 - 2 z.e ; running per-lane argmin ----
    float minv[MTILES][8]; int mini[MTILES][8];
#pragma unroll
    for (int u = 0; u < MTILES; ++u)
#pragma unroll
        for (int j = 0; j < 8; ++j) { minv[u][j] = 3.0e38f; mini[u][j] = 0; }

    for (int t = 0; t < 32; ++t) {
        const int k = t * 16 + m;                       // this lane's codebook column
        const _Float16* bh_base = eh + k * EH_STRIDE + g * 16;
        const _Float16* bl_base = el + k * EH_STRIDE + g * 16;
        v8f acc[MTILES];
#pragma unroll
        for (int u = 0; u < MTILES; ++u) acc[u] = (v8f){};
#pragma unroll
        for (int cc = 0; cc < 4; ++cc) {
            v16h bh = load_v16h(bh_base + cc * 32);
            v16h bl = load_v16h(bl_base + cc * 32);
#pragma unroll
            for (int u = 0; u < MTILES; ++u) {
                acc[u] = __builtin_amdgcn_wmma_f32_16x16x32_f16(false, ah[u][cc], false, bh,
                                                               (short)0, acc[u], false, false);
                acc[u] = __builtin_amdgcn_wmma_f32_16x16x32_f16(false, ah[u][cc], false, bl,
                                                               (short)0, acc[u], false, false);
                acc[u] = __builtin_amdgcn_wmma_f32_16x16x32_f16(false, al[u][cc], false, bh,
                                                               (short)0, acc[u], false, false);
            }
        }
        const float en = enorm[k];
#pragma unroll
        for (int u = 0; u < MTILES; ++u)
#pragma unroll
            for (int j = 0; j < 8; ++j) {
                float d = en - 2.0f * acc[u][j];
                if (d < minv[u][j]) { minv[u][j] = d; mini[u][j] = k; } // strict < keeps smallest k
            }
    }

    // ---- reduce (min,idx) across the 16 lanes holding each row ----
#pragma unroll
    for (int off = 1; off < 16; off <<= 1) {
#pragma unroll
        for (int u = 0; u < MTILES; ++u)
#pragma unroll
            for (int j = 0; j < 8; ++j) {
                float ov = __shfl_xor(minv[u][j], off, 32);
                int   oi = __shfl_xor(mini[u][j], off, 32);
                if (ov < minv[u][j] || (ov == minv[u][j] && oi < mini[u][j])) {
                    minv[u][j] = ov; mini[u][j] = oi;
                }
            }
    }
    if (m == 0) {
#pragma unroll
        for (int u = 0; u < MTILES; ++u)
#pragma unroll
            for (int j = 0; j < 8; ++j) {
                int row = wave * (16 * MTILES) + u * 16 + (g ? (j + 8) : j);
                idxs[row] = mini[u][j];
                idx_out[blockIdx.x * ROWS_PER_WG + row] = (float)mini[u][j];
            }
    }
    __syncthreads();

    // ---- gather exact fp32 z_q, write out (coalesced per channel), accumulate MSE ----
    float s = 0.f;
    for (int i = tid; i < ROWS_PER_WG * VQ_C; i += 256) {
        int r = i & (ROWS_PER_WG - 1), c = i >> 8;
        int nn = blockIdx.x * ROWS_PER_WG + r;
        int b2 = nn >> 12, p2 = nn & 4095;
        float zq = emb[idxs[r] * VQ_C + c];
        size_t off = ((size_t)b2 << 19) + ((size_t)c << 12) + (size_t)p2;
        float ze = z_e[off];
        zq_out[off] = zq;
        float d = zq - ze;
        s += d * d;
    }
    // fixed-order block reduction -> deterministic
#pragma unroll
    for (int off = 16; off >= 1; off >>= 1) s += __shfl_xor(s, off, 32);
    if (lane == 0) red[wave] = s;
    __syncthreads();
    if (tid == 0) {
        float tot = 0.f;
        for (int w = 0; w < 8; ++w) tot += red[w];
        partial[blockIdx.x] = tot;
    }
}

__global__ __launch_bounds__(256)
void vq_finalize(const float* __restrict__ partial, float* __restrict__ out) {
    __shared__ double red[256];
    double s = 0.0;
    for (int i = threadIdx.x; i < NUM_WGS; i += 256) s += (double)partial[i];
    red[threadIdx.x] = s;
    __syncthreads();
    for (int stride = 128; stride >= 1; stride >>= 1) {
        if (threadIdx.x < stride) red[threadIdx.x] += red[threadIdx.x + stride];
        __syncthreads();
    }
    if (threadIdx.x == 0) {
        float mean = (float)(red[0] / (double)NZ);
        out[NZ + 0] = mean;          // codebook_loss
        out[NZ + 1] = mean;          // commitment_loss (same forward value)
        out[NZ + 2] = 2.0f * mean;   // vq_loss, beta = 1
    }
}

extern "C" void kernel_launch(void* const* d_in, const int* in_sizes, int n_in,
                              void* d_out, int out_size, void* d_ws, size_t ws_size,
                              hipStream_t stream) {
    const float* z_e = (const float*)d_in[0];
    const float* emb = (const float*)d_in[1];
    float* out     = (float*)d_out;
    float* zq_out  = out;                    // [NZ] z_q_st == z_q forward
    float* idx_out = out + NZ + 3;           // [N] indices as float
    float* partial = (float*)d_ws;           // [NUM_WGS] per-block MSE partials

    vq_phase1<<<NUM_WGS, 256, LDS_BYTES, stream>>>(z_e, emb, zq_out, idx_out, partial);
    vq_finalize<<<1, 256, 0, stream>>>(partial, out);
}
